// MSA_45629732553422
// MI455X (gfx1250) — matrix operand order repaired and checked by reading, not compiled, not run on hardware
//
#include <hip/hip_runtime.h>
#include <hip/hip_bf16.h>

// ---- problem constants (match reference) ----
#define B_ 8
#define S_ 1024
#define H_ 16
#define Dh 64
#define HD (H_*Dh)   // 1024

typedef __bf16 bf16_t;
typedef bf16_t v16bf __attribute__((ext_vector_type(16)));
typedef float  v8f   __attribute__((ext_vector_type(8)));
typedef unsigned int u32x4 __attribute__((ext_vector_type(4)));
typedef int          i32x8 __attribute__((ext_vector_type(8)));
typedef int          i32x4 __attribute__((ext_vector_type(4)));

__device__ __forceinline__ v8f wmma_bf16(v16bf a, v16bf b, v8f c) {
    // D(f32 16x16) = A(bf16 16x32) * B(bf16 32x16) + C
    return __builtin_amdgcn_wmma_f32_16x16x32_bf16(
        /*neg_a=*/false, a, /*neg_b=*/false, b,
        /*c_mod=*/(short)0, c, /*reuse_a=*/false, /*reuse_b=*/false);
}

// A-fragment K index for element i of a 16x32 bf16 A tile (wave32 layout)
__device__ __forceinline__ int a_kidx(int i, int half) {
    return i + half * 16;
}

// ------------------------------------------------------------
// TDM: 1-D contiguous DMA of `nelem` bf16 from global to LDS.
// Builds the D# descriptor per CDNA5 ISA ch.8 (§8.3-8.6):
//   group0: count=1, lds_addr, global_addr[56:0], type=2
//   group1: data_size=2B, tensor_dim0=tile_dim0=nelem, stride=nelem
//   groups 2/3 (and trailing group operand): zero (tensor <= 2D)
// Toolchain here is clang-23/therock-10.0 -> 6-arg builtin form.
// ------------------------------------------------------------
__device__ __forceinline__ void tdm_load_1d(unsigned lds_off,
                                            const bf16_t* gsrc,
                                            int nelem) {
    const unsigned long long ga = (unsigned long long)(uintptr_t)gsrc;
    u32x4 g0;
    g0[0] = 1u;                                          // count=1 (valid), user mode
    g0[1] = lds_off;                                     // lds_addr (bytes)
    g0[2] = (unsigned)(ga & 0xffffffffull);              // global_addr[31:0]
    g0[3] = (unsigned)((ga >> 32) & 0x01ffffffull)       // global_addr[56:32]
          | 0x80000000u;                                 // type = 2 (bits 127:126)
    i32x8 g1;
    g1[0] = 0x00010000;                                  // wg_mask=0, data_size=1 (2 bytes)
    g1[1] = (nelem & 0xffff) << 16;                      // tensor_dim0[15:0] @ bits 63:48
    g1[2] = ((nelem >> 16) & 0xffff)                     // tensor_dim0[31:16]
          | (1 << 16);                                   // tensor_dim1 = 1
    g1[3] = (nelem & 0xffff) << 16;                      // tile_dim0 @ bits 127:112
    g1[4] = 0;                                           // tile_dim1=0, tile_dim2=0
    g1[5] = nelem;                                       // tensor_dim0_stride[31:0]
    g1[6] = 0;
    g1[7] = 0;
    const i32x4 gz4 = {0, 0, 0, 0};
    const i32x8 gz8 = {0, 0, 0, 0, 0, 0, 0, 0};
    __builtin_amdgcn_tensor_load_to_lds(g0, g1, gz4, gz4, gz8, /*cpol=*/0);
}

// ============================================================
// Kernel 1: per-head QKV projection.  One wave = 16 rows of S
// for one (b,h).  Q scaled by 1/sqrt(D); all outputs bf16.
// ============================================================
__global__ void msa_qkv_kernel(const float* __restrict__ x,
                               const float* __restrict__ Wq, const float* __restrict__ bq,
                               const float* __restrict__ Wk, const float* __restrict__ bk,
                               const float* __restrict__ Wv, const float* __restrict__ bv,
                               bf16_t* __restrict__ q_ws,
                               bf16_t* __restrict__ k_ws,
                               bf16_t* __restrict__ v_ws)
{
    const int lane = threadIdx.x & 31;
    const int wib  = threadIdx.x >> 5;
    const int gw   = blockIdx.x * (blockDim.x >> 5) + wib;   // 0 .. B*H*(S/16)-1
    const int t    = gw % (S_/16);
    const int bh   = gw / (S_/16);
    const int h    = bh % H_;
    const int b    = bh / H_;
    const int s0   = t * 16;

    const int am = lane & 15;     // row (A) / col (B,C)
    const int ah = lane >> 4;     // half selector

    // ---- load X tile as two A-fragments (K = 0..31, 32..63) ----
    v16bf a0, a1;
    const float* xrow = x + ((size_t)b * S_ + s0 + am) * HD + (size_t)h * Dh;
    #pragma unroll
    for (int i = 0; i < 16; ++i) {
        const int k = a_kidx(i, ah);
        a0[i] = (bf16_t)xrow[k];
        a1[i] = (bf16_t)xrow[k + 32];
    }

    // B-fragment of W^T: tileB[k][n] = W[h][n0+n][kbase+k]; contiguous in k per lane
    auto load_wfrag = [&](const float* W, int n0, int kbase) -> v16bf {
        v16bf r;
        const float* wrow = W + (size_t)h * Dh * Dh + (size_t)(n0 + am) * Dh + kbase + ah * 16;
        #pragma unroll
        for (int i = 0; i < 16; ++i) r[i] = (bf16_t)wrow[i];
        return r;
    };

    #pragma unroll
    for (int n0 = 0; n0 < Dh; n0 += 16) {
        v8f cq = {}, ck = {}, cv = {};
        cq = wmma_bf16(a0, load_wfrag(Wq, n0, 0),  cq);
        cq = wmma_bf16(a1, load_wfrag(Wq, n0, 32), cq);
        ck = wmma_bf16(a0, load_wfrag(Wk, n0, 0),  ck);
        ck = wmma_bf16(a1, load_wfrag(Wk, n0, 32), ck);
        cv = wmma_bf16(a0, load_wfrag(Wv, n0, 0),  cv);
        cv = wmma_bf16(a1, load_wfrag(Wv, n0, 32), cv);

        const float biasq = bq[h * Dh + n0 + am];
        const float biask = bk[h * Dh + n0 + am];
        const float biasv = bv[h * Dh + n0 + am];

        const size_t base = ((size_t)bh * S_ + s0) * Dh + n0 + am;
        #pragma unroll
        for (int r = 0; r < 8; ++r) {     // C layout: VGPR r holds row M = r + 8*half
            const size_t idx = base + (size_t)(r + 8 * ah) * Dh;
            q_ws[idx] = (bf16_t)((cq[r] + biasq) * 0.125f);   // fold 1/sqrt(64)
            k_ws[idx] = (bf16_t)(ck[r] + biask);
            v_ws[idx] = (bf16_t)(cv[r] + biasv);
        }
    }
}

// ============================================================
// Kernel 2: flash attention.  Block = 4 waves = 4 query tiles
// of the SAME (b,h); K/V tiles (32 keys x 64d, 4KB each, contiguous)
// are double-buffered into LDS by the Tensor Data Mover and
// shared by all 4 waves.  Online softmax; P bounced through
// LDS to build the A-fragment for the P*V WMMA.
// ============================================================
__global__ void msa_attn_kernel(const bf16_t* __restrict__ q_ws,
                                const bf16_t* __restrict__ k_ws,
                                const bf16_t* __restrict__ v_ws,
                                float* __restrict__ out)
{
    __shared__ bf16_t kt[2][32 * Dh];      // 2 x 4KB  K tiles
    __shared__ bf16_t vt[2][32 * Dh];      // 2 x 4KB  V tiles
    __shared__ bf16_t p_lds[4][16 * 32];   // per-wave P staging (1KB each)

    const int lane = threadIdx.x & 31;
    const int wib  = threadIdx.x >> 5;
    const int gw   = blockIdx.x * (blockDim.x >> 5) + wib;
    const int t    = gw % (S_/16);         // 4 consecutive t per block -> same bh
    const int bh   = gw / (S_/16);
    const int h    = bh % H_;
    const int b    = bh / H_;
    const int s0   = t * 16;

    const int am = lane & 15;
    const int ah = lane >> 4;

    // ---- Q A-fragments (direct from global; used 32x each) ----
    v16bf aq0, aq1;
    const bf16_t* qrow = q_ws + ((size_t)bh * S_ + s0 + am) * Dh;
    #pragma unroll
    for (int i = 0; i < 16; ++i) {
        const int k = a_kidx(i, ah);
        aq0[i] = qrow[k];
        aq1[i] = qrow[k + 32];
    }

    const bf16_t* Kbase = k_ws + (size_t)bh * S_ * Dh;
    const bf16_t* Vbase = v_ws + (size_t)bh * S_ * Dh;
    bf16_t* myP = &p_lds[wib][0];

    float mrow[8], lrow[8];
    v8f acc[4] = {};
    #pragma unroll
    for (int r = 0; r < 8; ++r) { mrow[r] = -__builtin_huge_valf(); lrow[r] = 0.0f; }

    constexpr int NT = S_ / 32;            // 32 key-tiles of 32 keys

    // prologue: DMA tile 0 into buffer 0 (one wave issues for the block)
    if (wib == 0) {
        tdm_load_1d((unsigned)(uintptr_t)&kt[0][0], Kbase, 32 * Dh);
        tdm_load_1d((unsigned)(uintptr_t)&vt[0][0], Vbase, 32 * Dh);
    }

    for (int j = 0; j < NT; ++j) {
        const int cur = j & 1;
        const int nxt = cur ^ 1;

        __syncthreads();   // all waves done reading buffer `nxt` (from iter j-1)
        if (wib == 0) {
            if (j + 1 < NT) {
                tdm_load_1d((unsigned)(uintptr_t)&kt[nxt][0],
                            Kbase + (size_t)(j + 1) * 32 * Dh, 32 * Dh);
                tdm_load_1d((unsigned)(uintptr_t)&vt[nxt][0],
                            Vbase + (size_t)(j + 1) * 32 * Dh, 32 * Dh);
                // TDM is in-order per wave: <=2 outstanding => tile j resident
                __builtin_amdgcn_s_wait_tensorcnt(2);
            } else {
                __builtin_amdgcn_s_wait_tensorcnt(0);
            }
        }
        __syncthreads();   // tile j visible to all waves

        const bf16_t* kb = &kt[cur][0];
        const bf16_t* vb = &vt[cur][0];

        // ---- scores for two 16-key subtiles: S = Q * K^T (B-frag from LDS) ----
        v8f sc[2];
        #pragma unroll
        for (int u = 0; u < 2; ++u) {
            const bf16_t* krow = kb + (size_t)(u * 16 + am) * Dh + ah * 16;
            const v16bf bk0 = *(const v16bf*)(krow);        // d = 0..31 slice
            const v16bf bk1 = *(const v16bf*)(krow + 32);   // d = 32..63 slice
            v8f s = {};
            s = wmma_bf16(aq0, bk0, s);
            s = wmma_bf16(aq1, bk1, s);
            sc[u] = s;
        }

        // ---- online softmax (row m = r + 8*half across 16 lanes of the half) ----
        #pragma unroll
        for (int r = 0; r < 8; ++r) {
            float v = fmaxf(sc[0][r], sc[1][r]);
            v = fmaxf(v, __shfl_xor(v, 1, 32));
            v = fmaxf(v, __shfl_xor(v, 2, 32));
            v = fmaxf(v, __shfl_xor(v, 4, 32));
            v = fmaxf(v, __shfl_xor(v, 8, 32));
            const float mnew  = fmaxf(mrow[r], v);
            const float p0    = __expf(sc[0][r] - mnew);
            const float p1    = __expf(sc[1][r] - mnew);
            const float scale = __expf(mrow[r] - mnew);   // 0 on first tile (exp(-inf))
            float rs = p0 + p1;
            rs += __shfl_xor(rs, 1, 32);
            rs += __shfl_xor(rs, 2, 32);
            rs += __shfl_xor(rs, 4, 32);
            rs += __shfl_xor(rs, 8, 32);
            lrow[r] = lrow[r] * scale + rs;
            mrow[r] = mnew;
            #pragma unroll
            for (int c4 = 0; c4 < 4; ++c4) acc[c4][r] *= scale;
            sc[0][r] = p0;
            sc[1][r] = p1;
        }

        // ---- C-layout P -> row-major LDS -> A-fragment (16x32) ----
        #pragma unroll
        for (int r = 0; r < 8; ++r) {
            const int m = r + 8 * ah;
            myP[m * 32 + am]      = (bf16_t)sc[0][r];
            myP[m * 32 + 16 + am] = (bf16_t)sc[1][r];
        }
        v16bf ap;
        const bf16_t* prow = myP + am * 32;
        #pragma unroll
        for (int i = 0; i < 16; ++i) ap[i] = prow[a_kidx(i, ah)];

        // ---- out += P * V over 4 output column tiles (B-frag from LDS) ----
        #pragma unroll
        for (int c4 = 0; c4 < 4; ++c4) {
            v16bf bv_;
            #pragma unroll
            for (int i = 0; i < 16; ++i) {
                const int key = i + 16 * ah;          // B-frag K index
                bv_[i] = vb[(size_t)key * Dh + c4 * 16 + am];
            }
            acc[c4] = wmma_bf16(ap, bv_, acc[c4]);
        }
    }

    // ---- finalize: divide by l, store fp32 [B,S,H*D] ----
    float* orow = out + ((size_t)b * S_ + s0) * HD + (size_t)h * Dh;
    #pragma unroll
    for (int c4 = 0; c4 < 4; ++c4) {
        #pragma unroll
        for (int r = 0; r < 8; ++r) {
            const int m = r + 8 * ah;
            orow[(size_t)m * HD + c4 * 16 + am] = acc[c4][r] / lrow[r];
        }
    }
}

// ============================================================
extern "C" void kernel_launch(void* const* d_in, const int* in_sizes, int n_in,
                              void* d_out, int out_size, void* d_ws, size_t ws_size,
                              hipStream_t stream) {
    const float* x  = (const float*)d_in[0];
    const float* Wq = (const float*)d_in[1];
    const float* bq = (const float*)d_in[2];
    const float* Wk = (const float*)d_in[3];
    const float* bk = (const float*)d_in[4];
    const float* Wv = (const float*)d_in[5];
    const float* bv = (const float*)d_in[6];
    float* out = (float*)d_out;

    const size_t qkv_elems = (size_t)B_ * H_ * S_ * Dh;   // 8M elems, bf16
    bf16_t* q_ws = (bf16_t*)d_ws;
    bf16_t* k_ws = q_ws + qkv_elems;
    bf16_t* v_ws = k_ws + qkv_elems;

    const int total_waves = B_ * H_ * (S_ / 16);          // 8192
    const int waves_per_block = 4;                        // 128 threads
    const int blocks = total_waves / waves_per_block;     // 2048

    msa_qkv_kernel<<<blocks, waves_per_block * 32, 0, stream>>>(
        x, Wq, bq, Wk, bk, Wv, bv, q_ws, k_ws, v_ws);
    msa_attn_kernel<<<blocks, waves_per_block * 32, 0, stream>>>(
        q_ws, k_ws, v_ws, out);
}
